// Channel_Transformer_WGCNLayer_89043261980865
// MI455X (gfx1250) — compile-verified, matrix operand order checked by reading, not run on hardware
//
#include <hip/hip_runtime.h>

typedef __attribute__((ext_vector_type(16))) _Float16 v16h;
typedef __attribute__((ext_vector_type(8)))  float    v8f;

union Frag { v16h v; uint4 q[2]; };

// ---------------- problem dims ----------------
constexpr int Bsz = 16;
constexpr int Nn  = 1024;   // graph nodes / attention feature dim h
constexpr int Co  = 512;    // channels (OUT_DIM == IN_DIM == 512)

// ---------------- GEMM tile config ----------------
constexpr int BM = 128, BN = 128, BK = 32;
constexpr int LDP = 40;    // padded LDS row (halfs), 80B stride keeps 16B align

__device__ __forceinline__ float sigmoid_f(float x) { return 1.0f / (1.0f + expf(-x)); }

// =====================================================================
// Tiled WMMA GEMM:  C[m,n] = scale * sum_k A[m,k] * Bt[n,k]  (+epilogue)
// A: row-major [M,K] f16,  Bt: row-major [Ncols,K] f16 (i.e. B transposed)
// =====================================================================
template<bool BIAS, bool LEAKY, bool RESID, bool TRANS, bool F16OUT>
__global__ __launch_bounds__(256) void gemm_wmma(
    const _Float16* __restrict__ A,  long long sAb,
    const _Float16* __restrict__ Bt, long long sBb,
    void* __restrict__ Out,          long long sOb,
    const float* __restrict__ bias,
    const float* __restrict__ resid, long long sRb,
    int M, int Ncols, int K, float scale)
{
    __shared__ alignas(16) _Float16 sA[BM][LDP];
    __shared__ alignas(16) _Float16 sB[BN][LDP];

    const int tid = threadIdx.x;
    const int bb  = blockIdx.z;
    const int m0  = blockIdx.y * BM;
    const int n0  = blockIdx.x * BN;

    const _Float16* Ab = A  + (size_t)bb * sAb;
    const _Float16* Bb = Bt + (size_t)bb * sBb;

    const int wid  = tid >> 5, lane = tid & 31;
    const int wm   = wid & 3,  wn   = wid >> 2;   // 4x2 wave grid
    const int ln   = lane & 15, hi  = lane >> 4;

    v8f acc[2][4] = {};

    for (int k0 = 0; k0 < K; k0 += BK) {
        __syncthreads();
        // Fill LDS: 128 rows x 32 halfs each for A-tile and Bt-tile
        #pragma unroll
        for (int i = 0; i < 2; ++i) {
            const int c  = i * 256 + tid;
            const int r  = c >> 2;
            const int kc = c & 3;
            *(uint4*)&sA[r][kc * 8] =
                *(const uint4*)(Ab + (size_t)(m0 + r) * K + k0 + kc * 8);
            *(uint4*)&sB[r][kc * 8] =
                *(const uint4*)(Bb + (size_t)(n0 + r) * K + k0 + kc * 8);
        }
        __syncthreads();

        // Fragment loads follow the ISA 16-bit A layout:
        // elem e in v16h -> K = (e&7) + 8*hi + 16*(e>>3)
        Frag af[2], bf[4];
        #pragma unroll
        for (int mt = 0; mt < 2; ++mt) {
            const int rr = wm * 32 + mt * 16 + ln;
            af[mt].q[0] = *(const uint4*)&sA[rr][hi * 8];
            af[mt].q[1] = *(const uint4*)&sA[rr][16 + hi * 8];
        }
        #pragma unroll
        for (int nt = 0; nt < 4; ++nt) {
            const int cc = wn * 64 + nt * 16 + ln;
            bf[nt].q[0] = *(const uint4*)&sB[cc][hi * 8];
            bf[nt].q[1] = *(const uint4*)&sB[cc][16 + hi * 8];
        }
        #pragma unroll
        for (int mt = 0; mt < 2; ++mt)
            #pragma unroll
            for (int nt = 0; nt < 4; ++nt)
                acc[mt][nt] = __builtin_amdgcn_wmma_f32_16x16x32_f16(
                    false, af[mt].v, false, bf[nt].v,
                    (short)0, acc[mt][nt], false, false);
    }

    // -------- epilogue --------
    const float* rb = RESID ? (resid + (size_t)bb * sRb) : nullptr;
    #pragma unroll
    for (int mt = 0; mt < 2; ++mt)
        #pragma unroll
        for (int nt = 0; nt < 4; ++nt) {
            const int gn = n0 + wn * 64 + nt * 16 + ln;
            #pragma unroll
            for (int r = 0; r < 8; ++r) {
                const int gm = m0 + wm * 32 + mt * 16 + hi * 8 + r;
                float v = acc[mt][nt][r] * scale;
                if constexpr (BIAS)  v += bias[gn];
                if constexpr (LEAKY) v = v > 0.f ? v : 0.01f * v;
                if constexpr (RESID) v += rb[(size_t)gm * Ncols + gn];
                const size_t oi = TRANS ? ((size_t)gn * M + gm)
                                        : ((size_t)gm * Ncols + gn);
                if constexpr (F16OUT)
                    ((_Float16*)Out)[(size_t)bb * sOb + oi] = (_Float16)v;
                else
                    ((float*)Out)[(size_t)bb * sOb + oi] = v;
            }
        }
}

// ================= elementwise / reduction kernels =================

__global__ void conv_f32_f16(const float* __restrict__ in,
                             _Float16* __restrict__ out, size_t n) {
    for (size_t i = (size_t)blockIdx.x * blockDim.x + threadIdx.x; i < n;
         i += (size_t)gridDim.x * blockDim.x)
        out[i] = (_Float16)in[i];
}

// one block per (b,n): d_inv_sqrt[b,n] = rsqrt(sum_m adj*sigmoid(edge))
__global__ __launch_bounds__(256) void rowsum_dinv(
    const float* __restrict__ adj, const float* __restrict__ edge,
    float* __restrict__ dinv) {
    __shared__ float s[256];
    const int row = blockIdx.x;          // b*Nn + n
    const int n   = row & (Nn - 1);
    const float* ar = adj  + (size_t)row * Nn;
    const float* er = edge + (size_t)n   * Nn;
    float sum = 0.f;
    for (int j = threadIdx.x; j < Nn; j += 256)
        sum += ar[j] * sigmoid_f(er[j]);
    s[threadIdx.x] = sum; __syncthreads();
    for (int off = 128; off > 0; off >>= 1) {
        if (threadIdx.x < off) s[threadIdx.x] += s[threadIdx.x + off];
        __syncthreads();
    }
    if (threadIdx.x == 0) {
        const float t = s[0];
        dinv[row] = (t > 0.f) ? rsqrtf(t) : 0.f;
    }
}

// WA16[b,n,m] = d[b,n]*adj[b,n,m]*sigmoid(edge[n,m])*d[b,m]
__global__ void wa_kernel(const float* __restrict__ adj,
                          const float* __restrict__ edge,
                          const float* __restrict__ dinv,
                          _Float16* __restrict__ wa) {
    const size_t total = (size_t)Bsz * Nn * Nn;
    for (size_t i = (size_t)blockIdx.x * blockDim.x + threadIdx.x; i < total;
         i += (size_t)gridDim.x * blockDim.x) {
        const int m = (int)(i & (Nn - 1));
        const int n = (int)((i >> 10) & (Nn - 1));
        const int b = (int)(i >> 20);
        const float v = dinv[b * Nn + n] * adj[i] *
                        sigmoid_f(edge[(size_t)n * Nn + m]) * dinv[b * Nn + m];
        wa[i] = (_Float16)v;
    }
}

// one block per row; population variance; optional in-place + f16 copy
__global__ __launch_bounds__(256) void layernorm_kernel(
    const float* __restrict__ in, float* __restrict__ out32,
    _Float16* __restrict__ out16, const float* __restrict__ gamma,
    const float* __restrict__ beta, int L) {
    __shared__ float s1[256], s2[256];
    const size_t base = (size_t)blockIdx.x * L;
    float sum = 0.f, sq = 0.f;
    for (int j = threadIdx.x; j < L; j += 256) {
        const float x = in[base + j];
        sum += x; sq += x * x;
    }
    s1[threadIdx.x] = sum; s2[threadIdx.x] = sq; __syncthreads();
    for (int off = 128; off > 0; off >>= 1) {
        if (threadIdx.x < off) {
            s1[threadIdx.x] += s1[threadIdx.x + off];
            s2[threadIdx.x] += s2[threadIdx.x + off];
        }
        __syncthreads();
    }
    const float mean = s1[0] / L;
    const float var  = s2[0] / L - mean * mean;
    const float rstd = rsqrtf(var + 1e-5f);
    for (int j = threadIdx.x; j < L; j += 256) {
        const float y = (in[base + j] - mean) * rstd * gamma[j] + beta[j];
        out32[base + j] = y;
        if (out16) out16[base + j] = (_Float16)y;
    }
}

// one block per row: softmax(f32 in) -> f16 out
__global__ __launch_bounds__(256) void softmax_kernel(
    const float* __restrict__ in, _Float16* __restrict__ out, int L) {
    __shared__ float s[256];
    const size_t base = (size_t)blockIdx.x * L;
    float mx = -3.4e38f;
    for (int j = threadIdx.x; j < L; j += 256) mx = fmaxf(mx, in[base + j]);
    s[threadIdx.x] = mx; __syncthreads();
    for (int off = 128; off > 0; off >>= 1) {
        if (threadIdx.x < off)
            s[threadIdx.x] = fmaxf(s[threadIdx.x], s[threadIdx.x + off]);
        __syncthreads();
    }
    mx = s[0]; __syncthreads();
    float sum = 0.f;
    for (int j = threadIdx.x; j < L; j += 256) sum += expf(in[base + j] - mx);
    s[threadIdx.x] = sum; __syncthreads();
    for (int off = 128; off > 0; off >>= 1) {
        if (threadIdx.x < off) s[threadIdx.x] += s[threadIdx.x + off];
        __syncthreads();
    }
    const float inv = 1.0f / s[0];
    for (int j = threadIdx.x; j < L; j += 256)
        out[base + j] = (_Float16)(expf(in[base + j] - mx) * inv);
}

// ================= host launch =================
extern "C" void kernel_launch(void* const* d_in, const int* in_sizes, int n_in,
                              void* d_out, int out_size, void* d_ws, size_t ws_size,
                              hipStream_t stream) {
    const float* node_feats = (const float*)d_in[0];
    const float* adj        = (const float*)d_in[1];
    const float* linear_w   = (const float*)d_in[2];
    const float* linear_b   = (const float*)d_in[3];
    const float* q_w        = (const float*)d_in[4];
    const float* q_b        = (const float*)d_in[5];
    const float* k_w        = (const float*)d_in[6];
    const float* k_b        = (const float*)d_in[7];
    const float* v_w        = (const float*)d_in[8];
    const float* v_b        = (const float*)d_in[9];
    const float* norm1_w    = (const float*)d_in[10];
    const float* norm1_b    = (const float*)d_in[11];
    const float* norm2_w    = (const float*)d_in[12];
    const float* norm2_b    = (const float*)d_in[13];
    const float* edge_w     = (const float*)d_in[14];

    char* w = (char*)d_ws;
    // ---- workspace layout (bytes), with lifetime-based aliasing ----
    constexpr size_t SZ_TX    = (size_t)Bsz * Co * Nn * 4;   // 33.5 MB f32
    constexpr size_t SZ_H16   = (size_t)Bsz * Co * Nn * 2;   // 16.8 MB f16
    constexpr size_t SZ_ATT   = (size_t)Bsz * Co * Co * 2;   //  8.4 MB f16
    constexpr size_t SZ_WA    = (size_t)Bsz * Nn * Nn * 2;   // 33.5 MB f16
    constexpr size_t SZ_X3    = (size_t)Bsz * Nn * Co * 4;   // 33.5 MB f32
    constexpr size_t SZ_DINV  = (size_t)Bsz * Nn * 4;

    size_t off = 0;
    float*     Tx    = (float*)(w + off);      off += SZ_TX;     // [b][c][n] f32 (post-LN1 in place)
    _Float16*  Txn16 = (_Float16*)(w + off);   off += SZ_H16;    // f16 copy of Tx
    _Float16*  k16   = (_Float16*)(w + off);   off += SZ_H16;
    _Float16*  vT16  = (_Float16*)(w + off);   off += SZ_H16;    // [b][h=1024][y=512]
    _Float16*  att16 = (_Float16*)(w + off);   off += SZ_ATT;    // [b][x][y]
    _Float16*  Tx2_16= (_Float16*)(w + off);   off += SZ_H16;    // [b][c][n] f16
    _Float16*  WA16  = (_Float16*)(w + off);   off += SZ_WA;
    float*     X3    = (float*)(w + off);                        // [b][n][c] f32
    float*     scores= (float*)(w + off);      off += SZ_X3;     // alias X3 (disjoint lifetime)
    float*     dinv  = (float*)(w + off);      off += SZ_DINV;
    _Float16*  nf16  = (_Float16*)(w + off);                     // node_feats f16
    _Float16*  q16   = (_Float16*)(w + off);   off += SZ_H16;    // alias nf16 (disjoint lifetime)
    _Float16*  lw16  = (_Float16*)(w + off);   off += (size_t)Co * Co * 2;
    _Float16*  qw16  = (_Float16*)(w + off);   off += (size_t)Nn * Nn * 2;
    _Float16*  kw16  = (_Float16*)(w + off);   off += (size_t)Nn * Nn * 2;
    _Float16*  vw16  = (_Float16*)(w + off);   off += (size_t)Nn * Nn * 2;

    const dim3 blk(256);

    // 1. f32 -> f16 conversions
    conv_f32_f16<<<2048, blk, 0, stream>>>(node_feats, nf16, (size_t)Bsz * Nn * Co);
    conv_f32_f16<<<512,  blk, 0, stream>>>(linear_w, lw16, (size_t)Co * Co);
    conv_f32_f16<<<1024, blk, 0, stream>>>(q_w, qw16, (size_t)Nn * Nn);
    conv_f32_f16<<<1024, blk, 0, stream>>>(k_w, kw16, (size_t)Nn * Nn);
    conv_f32_f16<<<1024, blk, 0, stream>>>(v_w, vw16, (size_t)Nn * Nn);

    // 2. degree norm + weighted adjacency (f16)
    rowsum_dinv<<<Bsz * Nn, blk, 0, stream>>>(adj, edge_w, dinv);
    wa_kernel<<<8192, blk, 0, stream>>>(adj, edge_w, dinv, WA16);

    // 3. X = leaky(nf @ lwT + b), stored transposed into Tx [b][c][n]
    gemm_wmma<true, true, false, true, false><<<dim3(Co/BN, Nn/BM, Bsz), blk, 0, stream>>>(
        nf16, (long long)Nn * Co, lw16, 0, Tx, (long long)Co * Nn,
        linear_b, nullptr, 0, Nn, Co, Co, 1.0f);

    // 4. LayerNorm over n (in place) + f16 copy
    layernorm_kernel<<<Bsz * Co, blk, 0, stream>>>(Tx, Tx, Txn16, norm1_w, norm1_b, Nn);

    // 5. q, k (bias), v (bias + leaky, stored transposed)
    gemm_wmma<true, false, false, false, true><<<dim3(Nn/BN, Co/BM, Bsz), blk, 0, stream>>>(
        Txn16, (long long)Co * Nn, qw16, 0, q16, (long long)Co * Nn,
        q_b, nullptr, 0, Co, Nn, Nn, 1.0f);
    gemm_wmma<true, false, false, false, true><<<dim3(Nn/BN, Co/BM, Bsz), blk, 0, stream>>>(
        Txn16, (long long)Co * Nn, kw16, 0, k16, (long long)Co * Nn,
        k_b, nullptr, 0, Co, Nn, Nn, 1.0f);
    gemm_wmma<true, true, false, true, true><<<dim3(Nn/BN, Co/BM, Bsz), blk, 0, stream>>>(
        Txn16, (long long)Co * Nn, vw16, 0, vT16, (long long)Nn * Co,
        v_b, nullptr, 0, Co, Nn, Nn, 1.0f);

    // 6. scores = (q @ kT) * N^-1/2   (f32)
    gemm_wmma<false, false, false, false, false><<<dim3(Co/BN, Co/BM, Bsz), blk, 0, stream>>>(
        q16, (long long)Co * Nn, k16, (long long)Co * Nn, scores, (long long)Co * Co,
        nullptr, nullptr, 0, Co, Co, Nn, 0.03125f);

    // 7. softmax rows -> att f16
    softmax_kernel<<<Bsz * Co, blk, 0, stream>>>(scores, att16, Co);

    // 8. Tx2 = Tx + att @ v   (residual in f32, stored f16 [b][c][n])
    gemm_wmma<false, false, true, false, true><<<dim3(Nn/BN, Co/BM, Bsz), blk, 0, stream>>>(
        att16, (long long)Co * Co, vT16, (long long)Nn * Co, Tx2_16, (long long)Co * Nn,
        nullptr, Tx, (long long)Co * Nn, Co, Nn, Co, 1.0f);

    // 9. X3 = W_A @ X2   (Bt operand is Tx2 directly: X2^T == Tx2)
    gemm_wmma<false, false, false, false, false><<<dim3(Co/BN, Nn/BM, Bsz), blk, 0, stream>>>(
        WA16, (long long)Nn * Nn, Tx2_16, (long long)Co * Nn, X3, (long long)Nn * Co,
        nullptr, nullptr, 0, Nn, Co, Nn, 1.0f);

    // 10. final LayerNorm over channels -> d_out
    layernorm_kernel<<<Bsz * Nn, blk, 0, stream>>>(X3, (float*)d_out, nullptr,
                                                   norm2_w, norm2_b, Co);
}
